// Attention_18081812316703
// MI455X (gfx1250) — matrix operand order repaired and checked by reading, not compiled, not run on hardware
//
#include <hip/hip_runtime.h>
#include <hip/hip_bf16.h>
#include <math.h>

// ---------------- problem constants ----------------
#define DIMC   384
#define NHEAD  12
#define HD     32          // head dim = DIMC/NHEAD
#define WSZ    7
#define TPW    49          // tokens per window
#define TPWP   64          // padded tokens per window
#define IMGB   32
#define IMGHW  56
#define TPI    (IMGHW*IMGHW)      // 3136 tokens per image
#define NTOK   (IMGB*TPI)         // 100352
#define NWIN   (IMGB*8*8)         // 2048 windows
#define QKVC   (3*DIMC)           // 1152

// LDS row strides (padded to dodge bank conflicts, 16B-aligned)
#define LDK32  40    // rows holding 32 bf16 -> 80B stride
#define LDK64  72    // rows holding 64 bf16 -> 144B stride
#define LDS65  65    // f32 score rows

typedef __attribute__((ext_vector_type(16))) __bf16 v16bf;
typedef __attribute__((ext_vector_type(8)))  __bf16 v8bf;
typedef __attribute__((ext_vector_type(4)))  __bf16 v4bf;
typedef __attribute__((ext_vector_type(8)))  float  v8f;
typedef unsigned int u32x4 __attribute__((ext_vector_type(4)));
typedef int          i32x8 __attribute__((ext_vector_type(8)));
typedef int          i32x4 __attribute__((ext_vector_type(4)));

__device__ __forceinline__ v16bf cat16(v8bf lo, v8bf hi) {
    return __builtin_shufflevector(lo, hi, 0,1,2,3,4,5,6,7,8,9,10,11,12,13,14,15);
}

__device__ __forceinline__ v4bf pack4(float4 v) {
    v4bf t;
    t[0] = (__bf16)v.x; t[1] = (__bf16)v.y; t[2] = (__bf16)v.z; t[3] = (__bf16)v.w;
    return t;
}

// A fragment (16x32 bf16, source row-major [m][k], row stride ldk).
// Lane L<16 -> M=L, K={0..7,16..23}; lane>=16 -> M=L-16, K={8..15,24..31}.
__device__ __forceinline__ v16bf load_frag_a(const __bf16* base, int ldk, int lane) {
    int m    = lane & 15;
    int koff = (lane >> 4) * 8;
    const __bf16* p = base + m * ldk;
    v8bf lo = *(const v8bf*)(p + koff);
    v8bf hi = *(const v8bf*)(p + 16 + koff);
    return cat16(lo, hi);
}

// B fragment (32x16 bf16) from an [n][k] LDS tile (n-major, row stride ldk).
// Lane L<16 -> N=L, K={0..7,16..23}; lane>=16 -> N=L-16, K={8..15,24..31}.
__device__ __forceinline__ v16bf load_frag_bT(const __bf16* base, int ldk, int lane) {
    int n    = lane & 15;
    int koff = (lane >> 4) * 8;
    const __bf16* p = base + n * ldk + koff;
    v8bf lo = *(const v8bf*)(p);
    v8bf hi = *(const v8bf*)(p + 16);
    return cat16(lo, hi);
}

#define WMMA_BF16(A, B, C) \
    __builtin_amdgcn_wmma_f32_16x16x32_bf16(false, (A), false, (B), (short)0, (C), false, false)

// ---------------------------------------------------------------------------
// TDM: copy one contiguous 49x32 bf16 window tile (3136B) global -> LDS,
// inserting 4 pad dwords after every 16 data dwords (LDS row stride 80B).
// D# layout per CDNA5 ISA.  6-arg builtin form.
// ---------------------------------------------------------------------------
#if __has_builtin(__builtin_amdgcn_tensor_load_to_lds) && __has_builtin(__builtin_amdgcn_s_wait_tensorcnt)
#define HAVE_TDM 1
__device__ __forceinline__ void tdm_load_win(const __bf16* gptr, unsigned lds_off) {
    unsigned long long ga = (unsigned long long)(uintptr_t)gptr;
    u32x4 g0;
    g0[0] = 1u;                                   // count=1 (valid user D#)
    g0[1] = lds_off;                              // lds_addr (bytes)
    g0[2] = (unsigned)(ga & 0xFFFFFFFFu);         // global_addr[31:0]
    g0[3] = (unsigned)((ga >> 32) & 0x01FFFFFFu)  // global_addr[56:32]
          | (2u << 30);                           // type=2 ("image")
    i32x8 g1;
    g1[0] = (1 << 16)      // data_size = 2 bytes
          | (1 << 20)      // pad_enable
          | (3 << 22)      // pad_interval: 16 dwords of data
          | (3 << 25);     // pad_amount:   4 dwords of pad
    g1[1] = (int)((1568u & 0xFFFFu) << 16);       // tensor_dim0 lo16 (bits 63:48)
    g1[2] = (int)(1u << 16);                      // tensor_dim0 hi16=0, tensor_dim1=1
    g1[3] = (int)((1568u & 0xFFFFu) << 16);       // tile_dim0 = 1568 elements
    g1[4] = 1;                                    // tile_dim1 = 1, tile_dim2 = 0
    g1[5] = 1568;                                 // tensor_dim0_stride lo32
    g1[6] = (int)((1568u & 0xFFFFu) << 16);       // stride0 hi16=0, stride1 lo16
    g1[7] = 0;
    i32x4 z4 = {0, 0, 0, 0};
    i32x8 z8 = {0, 0, 0, 0, 0, 0, 0, 0};
    __builtin_amdgcn_tensor_load_to_lds(g0, g1, z4, z4, z8, 0);
}
#else
#define HAVE_TDM 0
#endif

// ============================================================================
// Kernel 1: QKV GEMM [NTOK x 384] * [384 x 1152] with window-scatter epilogue.
// Block tile 128x128, 8 waves (wave tile 32x64), K stepped by 32 via LDS.
// ============================================================================
__global__ __launch_bounds__(256) void qkv_gemm(
        const float* __restrict__ x, const float* __restrict__ wqkv,
        __bf16* __restrict__ qbuf, __bf16* __restrict__ kbuf, __bf16* __restrict__ vbuf) {
    __shared__ alignas(16) __bf16 sA[128 * LDK32];   // [token][k]
    __shared__ alignas(16) __bf16 sB[128 * LDK32];   // [ochan][k]
    const int tid  = threadIdx.x;
    const int lane = tid & 31;
    const int wid  = tid >> 5;
    const int bm   = blockIdx.x;      // 0..783  (token tiles)
    const int bn   = blockIdx.y;      // 0..8    (out-channel tiles)
    const int wm   = wid >> 1;        // 0..3
    const int wn   = wid & 1;         // 0..1

    v8f acc[2][4] = {};

    for (int k0 = 0; k0 < DIMC; k0 += 32) {
        __syncthreads();
        // Phase 1: batch the global loads (all in flight before any use).
        // A: 4 x float4 = 4 consecutive tokens x 1 channel each.
        // (TPI % 4 == 0, so 4-aligned token groups never straddle an image.)
        float4 va[4];
        #pragma unroll
        for (int e = 0; e < 4; ++e) {
            int idx = e * 256 + tid;          // 0..1023
            int c   = idx >> 5;               // 0..31
            int t4  = (idx & 31) * 4;         // token group
            int tg  = bm * 128 + t4;
            int b   = tg / TPI;
            int pos = tg - b * TPI;
            va[e] = *(const float4*)&x[(size_t)(b * DIMC + (k0 + c)) * TPI + pos];
        }
        // B: 4 x float4 = 4 consecutive channels of one output row.
        float4 vb[4];
        #pragma unroll
        for (int e = 0; e < 4; ++e) {
            int idx = e * 256 + tid;          // 0..1023
            int o   = idx >> 3;               // 0..127
            int c4  = (idx & 7) * 4;          // 0..28
            vb[e] = *(const float4*)&wqkv[(size_t)(bn * 128 + o) * DIMC + k0 + c4];
        }
        // Phase 2: convert + LDS stores.
        #pragma unroll
        for (int e = 0; e < 4; ++e) {
            int idx = e * 256 + tid;
            int c   = idx >> 5;
            int t4  = (idx & 31) * 4;
            sA[(t4 + 0) * LDK32 + c] = (__bf16)va[e].x;
            sA[(t4 + 1) * LDK32 + c] = (__bf16)va[e].y;
            sA[(t4 + 2) * LDK32 + c] = (__bf16)va[e].z;
            sA[(t4 + 3) * LDK32 + c] = (__bf16)va[e].w;
        }
        #pragma unroll
        for (int e = 0; e < 4; ++e) {
            int idx = e * 256 + tid;
            int o   = idx >> 3;
            int c4  = (idx & 7) * 4;
            *(v4bf*)&sB[o * LDK32 + c4] = pack4(vb[e]);
        }
        // Prefetch next K-slice of the activation tile into cache.
        if (k0 + 32 < DIMC) {
            int c  = tid >> 3;            // 0..31
            int t  = (tid & 7) * 16;      // every 16 tokens (64B of fp32)
            int tg = bm * 128 + t;
            int b  = tg / TPI;
            int pos = tg - b * TPI;
            __builtin_prefetch(&x[(size_t)(b * DIMC + (k0 + 32 + c)) * TPI + pos], 0, 1);
        }
        __syncthreads();
        #pragma unroll
        for (int mt = 0; mt < 2; ++mt) {
            v16bf a = load_frag_a(sA + (wm * 32 + mt * 16) * LDK32, LDK32, lane);
            #pragma unroll
            for (int nt = 0; nt < 4; ++nt) {
                v16bf b = load_frag_bT(sB + (wn * 64 + nt * 16) * LDK32, LDK32, lane);
                acc[mt][nt] = WMMA_BF16(a, b, acc[mt][nt]);
            }
        }
    }

    // Epilogue: scatter to [win, head, tok, d] bf16; q pre-scaled by hd^-0.5.
    const float scale = 0.17677669529663687f;
    const int n     = lane & 15;
    const int mhalf = 8 * (lane >> 4);
    size_t tokpart[2][8];
    #pragma unroll
    for (int mt = 0; mt < 2; ++mt)
    #pragma unroll
    for (int r = 0; r < 8; ++r) {
        int tg  = bm * 128 + wm * 32 + mt * 16 + mhalf + r;
        int b   = tg / TPI;
        int pos = tg - b * TPI;
        int h   = pos / IMGHW, w2 = pos - h * IMGHW;
        int win = b * 64 + (h / WSZ) * 8 + (w2 / WSZ);
        int ti  = (h % WSZ) * WSZ + (w2 % WSZ);
        tokpart[mt][r] = (size_t)win * (NHEAD * TPW * HD) + (size_t)ti * HD;
    }
    #pragma unroll
    for (int nt = 0; nt < 4; ++nt) {
        int o     = bn * 128 + wn * 64 + nt * 16 + n;
        int which = o / DIMC;
        int cc    = o - which * DIMC;
        size_t headpart = (size_t)(cc >> 5) * (TPW * HD) + (cc & 31);
        __bf16* dst = (which == 0) ? qbuf : ((which == 1) ? kbuf : vbuf);
        float   sc  = (which == 0) ? scale : 1.0f;
        #pragma unroll
        for (int mt = 0; mt < 2; ++mt)
        #pragma unroll
        for (int r = 0; r < 8; ++r)
            dst[tokpart[mt][r] + headpart] = (__bf16)(acc[mt][nt][r] * sc);
    }
}

// ============================================================================
// Kernel 2: window attention. One block per (window, head); 4 waves.
// q/k tiles arrive via the Tensor Data Mover (with LDS row padding); v is
// loaded transposed by the waves. S = Q K^T (WMMA), softmax+bias, O = P V.
// ============================================================================
__global__ __launch_bounds__(128) void win_attn(
        const __bf16* __restrict__ qbuf, const __bf16* __restrict__ kbuf,
        const __bf16* __restrict__ vbuf, const float* __restrict__ bias_table,
        const int* __restrict__ rel_index, __bf16* __restrict__ abuf) {
    __shared__ alignas(16) __bf16 qs[TPWP * LDK32];   // [tok][d], stride 40
    __shared__ alignas(16) __bf16 ks[TPWP * LDK32];   // [tok][d], stride 40
    __shared__ alignas(16) __bf16 vt[HD * LDK64];     // [d][tok], stride 72
    __shared__ alignas(16) float  sS[TPWP * LDS65];   // scores, stride 65
    __shared__ alignas(16) __bf16 sP[TPWP * LDK64];   // probs, stride 72

    const int tid = threadIdx.x, lane = tid & 31, wid = tid >> 5;
    const int bw  = blockIdx.x;            // win*12 + head
    const int head = bw % NHEAD;
    const size_t base = (size_t)bw * TPW * HD;

#if HAVE_TDM
    // Wave 0 kicks off the two DMA tile loads (contiguous 3136B each, with
    // hardware-inserted LDS padding giving the 80B row stride).
    if (wid == 0) {
        tdm_load_win(qbuf + base, (unsigned)(uintptr_t)qs);
        tdm_load_win(kbuf + base, (unsigned)(uintptr_t)ks);
    }
    // Zero the pad rows 49..63 (disjoint from the TDM write region).
    for (int idx = tid; idx < (TPWP - TPW) * LDK32; idx += 128) {
        qs[TPW * LDK32 + idx] = (__bf16)0.f;
        ks[TPW * LDK32 + idx] = (__bf16)0.f;
    }
#else
    // Fallback: cooperative q/k loads.
    for (int idx = tid; idx < TPWP * HD; idx += 128) {
        int tok = idx >> 5, d = idx & 31;
        bool ok = tok < TPW;
        qs[tok * LDK32 + d] = ok ? qbuf[base + tok * HD + d] : (__bf16)0.f;
        ks[tok * LDK32 + d] = ok ? kbuf[base + tok * HD + d] : (__bf16)0.f;
    }
#endif
    // Load v (contiguous 1568 bf16) with two 16B vector loads per thread,
    // then scatter transposed into vt[d][tok].
    {
        v8bf vv[2];
        int cid0 = tid;            // chunks of 8 elements: 0..195 valid
        int cid1 = tid + 128;
        bool ok1 = cid1 < (TPW * HD) / 8;     // 196 chunks
        vv[0] = *(const v8bf*)&vbuf[base + cid0 * 8];
        if (ok1) vv[1] = *(const v8bf*)&vbuf[base + cid1 * 8];
        {
            int tok = (cid0 * 8) >> 5, d0 = (cid0 * 8) & 31;
            #pragma unroll
            for (int j = 0; j < 8; ++j) vt[(d0 + j) * LDK64 + tok] = vv[0][j];
        }
        if (ok1) {
            int tok = (cid1 * 8) >> 5, d0 = (cid1 * 8) & 31;
            #pragma unroll
            for (int j = 0; j < 8; ++j) vt[(d0 + j) * LDK64 + tok] = vv[1][j];
        }
        // Zero the pad columns tok 49..63 for all d.
        for (int idx = tid; idx < HD * (TPWP - TPW); idx += 128) {
            int d = idx / (TPWP - TPW), t = idx % (TPWP - TPW);
            vt[d * LDK64 + TPW + t] = (__bf16)0.f;
        }
    }
#if HAVE_TDM
    if (wid == 0) __builtin_amdgcn_s_wait_tensorcnt(0);
#endif
    __syncthreads();

    // S = Q K^T. Wave wid owns m-tile wid (rows wid*16..+15), K=32 = one WMMA.
    {
        v16bf a = load_frag_a(qs + wid * 16 * LDK32, LDK32, lane);
        #pragma unroll
        for (int nt = 0; nt < 4; ++nt) {
            v16bf b = load_frag_bT(ks + nt * 16 * LDK32, LDK32, lane);
            v8f c = {};
            c = WMMA_BF16(a, b, c);
            #pragma unroll
            for (int r = 0; r < 8; ++r) {
                int m = r + 8 * (lane >> 4), nn = lane & 15;
                sS[(wid * 16 + m) * LDS65 + nt * 16 + nn] = c[r];
            }
        }
    }
    __syncthreads();

    // Softmax with relative-position bias: one thread per row.
    if (tid < TPWP) {
        int i = tid;
        if (i < TPW) {
            const int* ri = rel_index + i * TPW;
            float mx = -1e30f;
            for (int j = 0; j < TPW; ++j) {
                float v = sS[i * LDS65 + j] + bias_table[ri[j] * NHEAD + head];
                sS[i * LDS65 + j] = v;
                mx = fmaxf(mx, v);
            }
            float sum = 0.f;
            for (int j = 0; j < TPW; ++j) {
                float e = __expf(sS[i * LDS65 + j] - mx);
                sS[i * LDS65 + j] = e;
                sum += e;
            }
            float inv = 1.0f / sum;
            for (int j = 0; j < TPW; ++j) sP[i * LDK64 + j] = (__bf16)(sS[i * LDS65 + j] * inv);
            for (int j = TPW; j < TPWP; ++j) sP[i * LDK64 + j] = (__bf16)0.f;
        } else {
            for (int j = 0; j < TPWP; ++j) sP[i * LDK64 + j] = (__bf16)0.f;
        }
    }
    __syncthreads();

    // O = P V : M=64 (wave per m-tile), N=32 (2 tiles), K=64 (2 steps of 32).
    v8f o0 = {}, o1 = {};
    #pragma unroll
    for (int kk = 0; kk < 2; ++kk) {
        v16bf a  = load_frag_a(sP + wid * 16 * LDK64 + kk * 32, LDK64, lane);
        v16bf b0 = load_frag_bT(vt + 0 * 16 * LDK64 + kk * 32, LDK64, lane);
        v16bf b1 = load_frag_bT(vt + 1 * 16 * LDK64 + kk * 32, LDK64, lane);
        o0 = WMMA_BF16(a, b0, o0);
        o1 = WMMA_BF16(a, b1, o1);
    }

    // Scatter to [token_global, 384] bf16 for the projection GEMM.
    const int win = bw / NHEAD;
    const int b_im = win >> 6, wl = win & 63, wh = wl >> 3, wwp = wl & 7;
    #pragma unroll
    for (int nt = 0; nt < 2; ++nt) {
        const v8f& o = nt ? o1 : o0;
        #pragma unroll
        for (int r = 0; r < 8; ++r) {
            int m = r + 8 * (lane >> 4), n = lane & 15;
            int tok = wid * 16 + m;
            if (tok < TPW) {
                int h  = wh * WSZ + tok / WSZ;
                int w2 = wwp * WSZ + tok % WSZ;
                size_t tg = (size_t)b_im * TPI + h * IMGHW + w2;
                abuf[tg * DIMC + head * HD + nt * 16 + n] = (__bf16)o[r];
            }
        }
    }
}

// ============================================================================
// Kernel 3: projection GEMM [NTOK x 384] * [384 x 384] + bias -> NCHW fp32.
// ============================================================================
__global__ __launch_bounds__(256) void proj_gemm(
        const __bf16* __restrict__ abuf, const float* __restrict__ wproj,
        const float* __restrict__ bproj, float* __restrict__ out) {
    __shared__ alignas(16) __bf16 sA[128 * LDK32];
    __shared__ alignas(16) __bf16 sB[128 * LDK32];
    const int tid  = threadIdx.x;
    const int lane = tid & 31;
    const int wid  = tid >> 5;
    const int bm   = blockIdx.x;   // 0..783
    const int bn   = blockIdx.y;   // 0..2
    const int wm   = wid >> 1;
    const int wn   = wid & 1;

    v8f acc[2][4] = {};

    for (int k0 = 0; k0 < DIMC; k0 += 32) {
        __syncthreads();
        // A: two 16B loads per thread (bf16 source, no conversion needed).
        u32x4 ua[2];
        #pragma unroll
        for (int e = 0; e < 2; ++e) {
            int idx = e * 256 + tid;          // 0..511
            int t   = idx >> 2;               // 0..127
            int c8  = (idx & 3) * 8;          // 0,8,16,24
            ua[e] = *(const u32x4*)&abuf[(size_t)(bm * 128 + t) * DIMC + k0 + c8];
        }
        // B: 4 x float4 per thread.
        float4 vb[4];
        #pragma unroll
        for (int e = 0; e < 4; ++e) {
            int idx = e * 256 + tid;
            int o   = idx >> 3;
            int c4  = (idx & 7) * 4;
            vb[e] = *(const float4*)&wproj[(size_t)(bn * 128 + o) * DIMC + k0 + c4];
        }
        #pragma unroll
        for (int e = 0; e < 2; ++e) {
            int idx = e * 256 + tid;
            int t   = idx >> 2;
            int c8  = (idx & 3) * 8;
            *(u32x4*)&sA[t * LDK32 + c8] = ua[e];
        }
        #pragma unroll
        for (int e = 0; e < 4; ++e) {
            int idx = e * 256 + tid;
            int o   = idx >> 3;
            int c4  = (idx & 7) * 4;
            *(v4bf*)&sB[o * LDK32 + c4] = pack4(vb[e]);
        }
        if (k0 + 32 < DIMC) {
            int t = tid >> 1;                       // 0..127
            int c = (tid & 1) * 16;                 // 2 x 32B
            __builtin_prefetch(&abuf[(size_t)(bm * 128 + t) * DIMC + k0 + 32 + c], 0, 1);
        }
        __syncthreads();
        #pragma unroll
        for (int mt = 0; mt < 2; ++mt) {
            v16bf a = load_frag_a(sA + (wm * 32 + mt * 16) * LDK32, LDK32, lane);
            #pragma unroll
            for (int nt = 0; nt < 4; ++nt) {
                v16bf b = load_frag_bT(sB + (wn * 64 + nt * 16) * LDK32, LDK32, lane);
                acc[mt][nt] = WMMA_BF16(a, b, acc[mt][nt]);
            }
        }
    }

    // out[(b*DIMC + o)*TPI + pos] = tokpart (b*DIMC*TPI + pos) + o*TPI
    const int n     = lane & 15;
    const int mhalf = 8 * (lane >> 4);
    size_t tokpart[2][8];
    #pragma unroll
    for (int mt = 0; mt < 2; ++mt)
    #pragma unroll
    for (int r = 0; r < 8; ++r) {
        int tg  = bm * 128 + wm * 32 + mt * 16 + mhalf + r;
        int b   = tg / TPI;
        int pos = tg - b * TPI;
        tokpart[mt][r] = (size_t)b * DIMC * TPI + pos;
    }
    #pragma unroll
    for (int nt = 0; nt < 4; ++nt) {
        int o = bn * 128 + wn * 64 + nt * 16 + n;
        size_t opart = (size_t)o * TPI;
        float  bias  = bproj[o];
        #pragma unroll
        for (int mt = 0; mt < 2; ++mt)
        #pragma unroll
        for (int r = 0; r < 8; ++r)
            out[tokpart[mt][r] + opart] = acc[mt][nt][r] + bias;
    }
}

// ============================================================================
extern "C" void kernel_launch(void* const* d_in, const int* in_sizes, int n_in,
                              void* d_out, int out_size, void* d_ws, size_t ws_size,
                              hipStream_t stream) {
    const float* x     = (const float*)d_in[0];
    const float* wqkv  = (const float*)d_in[1];
    const float* wproj = (const float*)d_in[2];
    const float* bproj = (const float*)d_in[3];
    const float* btab  = (const float*)d_in[4];
    const int*   ridx  = (const int*)d_in[5];

    const size_t E = (size_t)NWIN * NHEAD * TPW * HD;   // 38,535,168 elements
    __bf16* qb = (__bf16*)d_ws;
    __bf16* kb = qb + E;
    __bf16* vb = kb + E;
    __bf16* ab = vb + E;

    qkv_gemm<<<dim3(NTOK / 128, QKVC / 128), 256, 0, stream>>>(x, wqkv, qb, kb, vb);
    win_attn<<<dim3(NWIN * NHEAD), 128, 0, stream>>>(qb, kb, vb, btab, ridx, ab);
    proj_gemm<<<dim3(NTOK / 128, DIMC / 128), 256, 0, stream>>>(ab, wproj, bproj, (float*)d_out);
}
